// GIN_15685220565565
// MI455X (gfx1250) — matrix operand order repaired and checked by reading, not compiled, not run on hardware
//
#include <hip/hip_runtime.h>
#include <hip/hip_bf16.h>

// ---------------------------------------------------------------------------
// GIN GNN + edge classifier for MI455X (gfx1250), wave32 + WMMA f16->f32.
//
// N = 100000 nodes, E = 1600000 edges, NF=32, EF=16, H=64.
// N % 16 == 0, E % 64 == 0 -> all tiles full.
//
// GEMMs on v_wmma_f32_16x16x32_f16 (f32 accumulate). Weights pre-transposed
// and K-padded to f16 Wt[outCol][Kpad] so a lane's B fragment is one
// contiguous 32B load; B fragments are held in registers and reused across
// 4 row tiles in the edge classifier. Edge-classifier input rows (x[src],
// x[dst] as f16) are gathered with GLOBAL_LOAD_ASYNC_TO_LDS_B128 (ASYNCcnt).
// Eval BatchNorm is folded into classifier weights/biases. Aggregation uses
// f32 global atomics (x is 25.6 MB -> L2 resident).
// ---------------------------------------------------------------------------

typedef __attribute__((ext_vector_type(16))) _Float16 v16h;
typedef __attribute__((ext_vector_type(8)))  _Float16 v8h;
typedef __attribute__((ext_vector_type(8)))  float    v8f;

#define N_NODES 100000
#define N_EDGES 1600000
#define HDIM    64
#define BN_INV  0.9999950000374997f   /* 1/sqrt(1 + 1e-5) */

// Generic pointer -> LDS byte offset (addrspace(3)) for async-to-LDS asm.
typedef __attribute__((address_space(3))) const void lds_cvoid;
__device__ __forceinline__ unsigned lds_offset(const void* p) {
  return (unsigned)(unsigned long long)(lds_cvoid*)p;
}

// ---------------------------------------------------------------------------
// WMMA fragment loaders (16-bit k order used here is contiguous, k = 16h + s, same for A and B; not AMD's table order):
//  A 16x32: lane&15 = M; lanes 0-15 hold K 0..15, lanes 16-31 hold
//           K 16..31; element i -> VGPR i/2, half i%2.
//  B 32x16: lane&15 = N; lanes 0-15 hold K 0..15, lanes 16-31 K 16..31.
//  C/D:     lane&15 = N; VGPR r -> M = r (lanes 0-15) or r+8 (lanes 16-31).
// ---------------------------------------------------------------------------
__device__ __forceinline__ v16h load_a_frag_lds(const _Float16* lds, int lda,
                                                int k0, int lane) {
  const int m    = lane & 15;
  const int base = (lane < 16) ? 0 : 16;
  const _Float16* p = lds + m * lda + k0 + base;
  v8h lo = *(const v8h*)(p);        // K = k0+base .. +7
  v8h hi = *(const v8h*)(p + 8);    // K = k0+base+8 .. +15
  v16h a;
#pragma unroll
  for (int i = 0; i < 8; ++i) { a[i] = lo[i]; a[i + 8] = hi[i]; }
  return a;
}

__device__ __forceinline__ v16h load_b_frag(const _Float16* __restrict__ wt,
                                            int kpad, int colBase, int k0,
                                            int lane) {
  const int n  = lane & 15;
  const int kb = (lane < 16) ? 0 : 16;
  const _Float16* p = wt + (size_t)(colBase + n) * kpad + k0 + kb;
  return *(const v16h*)p;           // 32B contiguous, 32B aligned
}

__device__ __forceinline__ void store_acc_relu_f16(
    v8f c, const float* __restrict__ bias, int colBase, int lane,
    _Float16* out, int ldo) {
  const int n = lane & 15, rbase = (lane < 16) ? 0 : 8;
  const float bv = bias[colBase + n];
#pragma unroll
  for (int r = 0; r < 8; ++r) {
    float v = fmaxf(c[r] + bv, 0.0f);
    out[(rbase + r) * ldo + colBase + n] = (_Float16)v;
  }
}

// 16x16 output tile: C = A(16xK)*B + bias, ReLU, f16 -> LDS.
template <int KSTEPS>
__device__ __forceinline__ void gemm_tile_to_lds(
    const _Float16* ldsA, int lda, const _Float16* __restrict__ wt, int kpad,
    const float* __restrict__ bias, int colBase, int lane,
    _Float16* ldsOut, int ldo) {
  v8f c = {};
#pragma unroll
  for (int s = 0; s < KSTEPS; ++s) {
    v16h a = load_a_frag_lds(ldsA, lda, s * 32, lane);
    v16h b = load_b_frag(wt, kpad, colBase, s * 32, lane);
    c = __builtin_amdgcn_wmma_f32_16x16x32_f16(false, a, false, b,
                                               (short)0, c, false, false);
  }
  store_acc_relu_f16(c, bias, colBase, lane, ldsOut, ldo);
}

// 16x16 output tile: + bias, ReLU, f32 -> global (optional f16 shadow copy).
template <int KSTEPS>
__device__ __forceinline__ void gemm_tile_to_gmem(
    const _Float16* ldsA, int lda, const _Float16* __restrict__ wt, int kpad,
    const float* __restrict__ bias, int colBase, int lane,
    float* __restrict__ gout, _Float16* __restrict__ gout16,
    int ldg, long row0) {
  v8f c = {};
#pragma unroll
  for (int s = 0; s < KSTEPS; ++s) {
    v16h a = load_a_frag_lds(ldsA, lda, s * 32, lane);
    v16h b = load_b_frag(wt, kpad, colBase, s * 32, lane);
    c = __builtin_amdgcn_wmma_f32_16x16x32_f16(false, a, false, b,
                                               (short)0, c, false, false);
  }
  const int n = lane & 15, rbase = (lane < 16) ? 0 : 8;
  const float bv = bias[colBase + n];
#pragma unroll
  for (int r = 0; r < 8; ++r) {
    float v = fmaxf(c[r] + bv, 0.0f);
    const size_t idx = (row0 + rbase + r) * ldg + colBase + n;
    gout[idx] = v;
    if (gout16) gout16[idx] = (_Float16)v;
  }
}

// ---------------------------------------------------------------------------
// Weight prep: W[K,Ncols] f32 -> Wt[Npad][Kpad] f16 transposed, zero padded.
// Optionally folds eval BN: W' = W*gamma*inv, b' = b*gamma*inv + beta.
// ---------------------------------------------------------------------------
__global__ void __launch_bounds__(256) prep_weight_kernel(
    const float* __restrict__ W, const float* __restrict__ bias,
    const float* __restrict__ gamma, const float* __restrict__ beta,
    _Float16* __restrict__ Wt, float* __restrict__ bOut,
    int K, int Ncols, int Kpad, int Npad) {
  const int idx = blockIdx.x * 256 + threadIdx.x;
  const int total = Npad * Kpad;
  if (idx < total) {
    const int col = idx / Kpad, k = idx % Kpad;
    float v = 0.0f;
    if (k < K && col < Ncols) {
      v = W[(size_t)k * Ncols + col];
      if (gamma) v *= gamma[col] * BN_INV;
    }
    Wt[idx] = (_Float16)v;
  }
  if (idx < Npad) {
    float b = 0.0f;
    if (idx < Ncols) {
      b = bias[idx];
      if (gamma) b = b * gamma[idx] * BN_INV + beta[idx];
    }
    bOut[idx] = b;
  }
}

__global__ void __launch_bounds__(256) zero_kernel(float* __restrict__ p, int n) {
  const int i = blockIdx.x * 256 + threadIdx.x;
  if (i < n) p[i] = 0.0f;
}

// ---------------------------------------------------------------------------
// Node encoder: x = relu(nf[N,32] @ Wenc[32,64] + b).  16 nodes per block.
// ---------------------------------------------------------------------------
__global__ void __launch_bounds__(256) enc_kernel(
    const float* __restrict__ nf, const _Float16* __restrict__ wt,
    const float* __restrict__ bias, float* __restrict__ x) {
  __shared__ __align__(32) _Float16 aT[16 * 32];
  const int t = threadIdx.x;
  const long row0 = (long)blockIdx.x * 16;
  for (int i = t; i < 16 * 32; i += 256) {
    const int r = i >> 5, c = i & 31;
    aT[i] = (_Float16)nf[(row0 + r) * 32 + c];
  }
  __syncthreads();
  const int w = t >> 5, lane = t & 31;
  if (w < 4)
    gemm_tile_to_gmem<1>(aT, 32, wt, 32, bias, w * 16, lane,
                         x, nullptr, HDIM, row0);
}

// ---------------------------------------------------------------------------
// Aggregation: agg[dst] += x[src] (f32 atomics; x and agg are L2-resident).
// One thread per (edge, 4-float chunk).
// ---------------------------------------------------------------------------
__global__ void __launch_bounds__(256) scatter_kernel(
    const float* __restrict__ x, const int* __restrict__ src,
    const int* __restrict__ dst, float* __restrict__ agg) {
  const long tid = (long)blockIdx.x * 256 + threadIdx.x;
  const long e = tid >> 4;
  if (e >= N_EDGES) return;
  const int c4 = (int)(tid & 15) << 2;
  const int s = src[e], d = dst[e];
  const float4 v = *(const float4*)(x + (size_t)s * HDIM + c4);
  float* a = agg + (size_t)d * HDIM + c4;
  atomicAdd(a + 0, v.x);
  atomicAdd(a + 1, v.y);
  atomicAdd(a + 2, v.z);
  atomicAdd(a + 3, v.w);
}

// ---------------------------------------------------------------------------
// GIN MLP (fused, in-place on x):
//   h  = (1+eps)*x + agg              [16,64]  f32->f16 LDS
//   o1 = relu(h @ W1[64,128] + b1)    [16,128] f16 LDS
//   x  = relu(o1 @ W2[128,64] + b2)   f32 global (+ f16 shadow on last layer)
// ---------------------------------------------------------------------------
__global__ void __launch_bounds__(256) gin_mlp_kernel(
    float* __restrict__ x, _Float16* __restrict__ x16,
    const float* __restrict__ agg, const float* __restrict__ epsP,
    const _Float16* __restrict__ wt1, const float* __restrict__ b1,
    const _Float16* __restrict__ wt2, const float* __restrict__ b2) {
  __shared__ __align__(32) _Float16 hT[16 * 64];
  __shared__ __align__(32) _Float16 o1[16 * 128];
  const int t = threadIdx.x;
  const long row0 = (long)blockIdx.x * 16;
  const float opE = 1.0f + epsP[0];
  for (int i = t; i < 16 * 64; i += 256) {
    const int r = i >> 6, c = i & 63;
    const size_t idx = (row0 + r) * HDIM + c;
    hT[i] = (_Float16)(opE * x[idx] + agg[idx]);
  }
  __syncthreads();
  const int w = t >> 5, lane = t & 31;
  gemm_tile_to_lds<2>(hT, 64, wt1, 64, b1, w * 16, lane, o1, 128);
  __syncthreads();
  if (w < 4)
    gemm_tile_to_gmem<4>(o1, 128, wt2, 128, b2, w * 16, lane,
                         x, x16, HDIM, row0);
}

// ---------------------------------------------------------------------------
// Edge classifier: 64 edges per block (4 row tiles); B fragments reused in
// registers across row tiles. x-row gathers via async global->LDS B128.
//   e  = [x16[src] | x16[dst] | edge_attr | 0]  [64,160] f16 LDS
//   o1 = relu(e @ W0'[160,128] + b0')           (BN folded)
//   o2 = relu(o1 @ W1'[128,64] + b1')
//   out= o2 @ W2[64,2] + b2                     (W2 zero-padded to 16 cols)
// ---------------------------------------------------------------------------
#define ET_OFF 0
#define O1_OFF (64 * 160 * 2)             /* 20480 */
#define O2_OFF (O1_OFF + 64 * 128 * 2)    /* 36864 */
#define SMEM_BYTES (O2_OFF + 64 * 64 * 2) /* 45056 */

__global__ void __launch_bounds__(256) edge_cls_kernel(
    const _Float16* __restrict__ x16, const float* __restrict__ eattr,
    const int* __restrict__ src, const int* __restrict__ dst,
    const _Float16* __restrict__ w0, const float* __restrict__ b0,
    const _Float16* __restrict__ w1, const float* __restrict__ b1,
    const _Float16* __restrict__ w2, const float* __restrict__ b2,
    float* __restrict__ out) {
  __shared__ __align__(32) unsigned char smem[SMEM_BYTES];
  _Float16* eT = (_Float16*)(smem + ET_OFF);
  _Float16* o1 = (_Float16*)(smem + O1_OFF);
  _Float16* o2 = (_Float16*)(smem + O2_OFF);
  const int t = threadIdx.x;
  const long e0 = (long)blockIdx.x * 64;

  // ---- stage x16[src]/x16[dst] rows: 64 rows x 2 sides x 8 x 16B chunks
  //      = 1024 async B128 copies (ASYNCcnt), 4 per thread.
#pragma unroll
  for (int j = 0; j < 4; ++j) {
    const int id   = t + 256 * j;      // 0..1023
    const int side = id >> 9;          // 0 = src, 1 = dst
    const int row  = (id >> 3) & 63;
    const int ch   = id & 7;
    const int node = side ? dst[e0 + row] : src[e0 + row];
    const _Float16* g = x16 + (size_t)node * HDIM + ch * 8;
    const unsigned ldsOff = lds_offset(eT + row * 160 + side * 64 + ch * 8);
    asm volatile("global_load_async_to_lds_b128 %0, %1, off"
                 :: "v"(ldsOff), "v"(g) : "memory");
  }
  // ---- edge_attr (f32->f16) + zero K padding, cols 128..159
#pragma unroll
  for (int j = 0; j < 8; ++j) {
    const int id = t + 256 * j;        // 0..2047
    const int row = id >> 5, c = id & 31;
    const float v = (c < 16) ? eattr[(e0 + row) * 16 + c] : 0.0f;
    eT[row * 160 + 128 + c] = (_Float16)v;
  }
  asm volatile("s_wait_asynccnt 0x0" ::: "memory");
  __syncthreads();

  const int w = t >> 5, lane = t & 31;
  // ---- Layer 0: 8 waves = 8 col tiles; K=160 (5 k-steps); 4 row tiles.
  {
    v8f acc[4] = {};
#pragma unroll
    for (int s = 0; s < 5; ++s) {
      v16h b = load_b_frag(w0, 160, w * 16, s * 32, lane);
#pragma unroll
      for (int rt = 0; rt < 4; ++rt) {
        v16h a = load_a_frag_lds(eT + rt * 16 * 160, 160, s * 32, lane);
        acc[rt] = __builtin_amdgcn_wmma_f32_16x16x32_f16(
            false, a, false, b, (short)0, acc[rt], false, false);
      }
    }
#pragma unroll
    for (int rt = 0; rt < 4; ++rt)
      store_acc_relu_f16(acc[rt], b0, w * 16, lane, o1 + rt * 16 * 128, 128);
  }
  __syncthreads();
  // ---- Layer 1: 8 waves = 4 col tiles x 2 row-tile pairs; K=128 (4 k-steps).
  {
    const int ct = (w & 3) * 16, rb = (w >> 2) * 2;
    v8f acc[2] = {};
#pragma unroll
    for (int s = 0; s < 4; ++s) {
      v16h b = load_b_frag(w1, 128, ct, s * 32, lane);
#pragma unroll
      for (int k = 0; k < 2; ++k) {
        v16h a = load_a_frag_lds(o1 + (rb + k) * 16 * 128, 128, s * 32, lane);
        acc[k] = __builtin_amdgcn_wmma_f32_16x16x32_f16(
            false, a, false, b, (short)0, acc[k], false, false);
      }
    }
#pragma unroll
    for (int k = 0; k < 2; ++k)
      store_acc_relu_f16(acc[k], b1, ct, lane, o2 + (rb + k) * 16 * 64, 64);
  }
  __syncthreads();
  // ---- Layer 2: 4 waves, one row tile each; 2 real outputs (padded to 16).
  if (w < 4) {
    v8f c = {};
#pragma unroll
    for (int s = 0; s < 2; ++s) {
      v16h a = load_a_frag_lds(o2 + w * 16 * 64, 64, s * 32, lane);
      v16h b = load_b_frag(w2, 64, 0, s * 32, lane);
      c = __builtin_amdgcn_wmma_f32_16x16x32_f16(false, a, false, b,
                                                 (short)0, c, false, false);
    }
    const int n = lane & 15, rbase = (lane < 16) ? 0 : 8;
    if (n < 2) {
      const float bv = b2[n];
#pragma unroll
      for (int r = 0; r < 8; ++r)
        out[(e0 + w * 16 + rbase + r) * 2 + n] = c[r] + bv;
    }
  }
}

// ---------------------------------------------------------------------------
// Host launcher.
// d_in order: setup_inputs() dict order {node_features, edge_attr, params,
// edge_index}; params flattened as JAX pytree leaves (dict keys sorted):
//   cls[0]{b,beta,gamma,w}, cls[1]{b,beta,gamma,w}, cls[2]{b,w},
//   enc_b, enc_w, gin[0..2]{b1,b2,eps,w1,w2}
// ---------------------------------------------------------------------------
enum {
  IN_NF = 0, IN_EATTR = 1,
  IN_C0_B = 2, IN_C0_BETA, IN_C0_GAMMA, IN_C0_W,
  IN_C1_B,     IN_C1_BETA, IN_C1_GAMMA, IN_C1_W,
  IN_C2_B,     IN_C2_W,
  IN_ENC_B,    IN_ENC_W,
  IN_G0_B1, IN_G0_B2, IN_G0_EPS, IN_G0_W1, IN_G0_W2,
  IN_G1_B1, IN_G1_B2, IN_G1_EPS, IN_G1_W1, IN_G1_W2,
  IN_G2_B1, IN_G2_B2, IN_G2_EPS, IN_G2_W1, IN_G2_W2
};

extern "C" void kernel_launch(void* const* d_in, const int* in_sizes, int n_in,
                              void* d_out, int out_size, void* d_ws, size_t ws_size,
                              hipStream_t stream) {
  const float* nf    = (const float*)d_in[IN_NF];
  const float* eattr = (const float*)d_in[IN_EATTR];
  const int* eidx    = (const int*)d_in[n_in - 1];  // edge_index is last
  const int* src = eidx;
  const int* dst = eidx + N_EDGES;

  // ---- workspace carve-out (256B aligned) ----
  size_t off = 0;
  auto alloc = [&](size_t bytes) -> void* {
    off = (off + 255) & ~(size_t)255;
    void* p = (char*)d_ws + off;
    off += bytes;
    return p;
  };
  float*    x   = (float*)alloc((size_t)N_NODES * HDIM * 4);
  float*    agg = (float*)alloc((size_t)N_NODES * HDIM * 4);
  _Float16* x16 = (_Float16*)alloc((size_t)N_NODES * HDIM * 2);
  _Float16* wtEnc = (_Float16*)alloc(64 * 32 * 2);   float* bEnc = (float*)alloc(64 * 4);
  _Float16* wtG1[3]; float* bG1[3];
  _Float16* wtG2[3]; float* bG2[3];
  for (int L = 0; L < 3; ++L) {
    wtG1[L] = (_Float16*)alloc(128 * 64 * 2);  bG1[L] = (float*)alloc(128 * 4);
    wtG2[L] = (_Float16*)alloc(64 * 128 * 2);  bG2[L] = (float*)alloc(64 * 4);
  }
  _Float16* wtC0 = (_Float16*)alloc(128 * 160 * 2); float* bC0 = (float*)alloc(128 * 4);
  _Float16* wtC1 = (_Float16*)alloc(64 * 128 * 2);  float* bC1 = (float*)alloc(64 * 4);
  _Float16* wtC2 = (_Float16*)alloc(16 * 64 * 2);   float* bC2 = (float*)alloc(16 * 4);
  (void)ws_size; (void)in_sizes; (void)out_size;

  // ---- weight preprocessing ----
  auto prep = [&](int iW, int iB, int iG, int iBeta, _Float16* wt, float* b,
                  int K, int Nc, int Kp, int Np) {
    const int total = Np * Kp;
    prep_weight_kernel<<<(total + 255) / 256, 256, 0, stream>>>(
        (const float*)d_in[iW], (const float*)d_in[iB],
        iG >= 0 ? (const float*)d_in[iG] : nullptr,
        iBeta >= 0 ? (const float*)d_in[iBeta] : nullptr,
        wt, b, K, Nc, Kp, Np);
  };
  prep(IN_ENC_W, IN_ENC_B, -1, -1, wtEnc, bEnc, 32, 64, 32, 64);
  const int gW1[3]  = {IN_G0_W1,  IN_G1_W1,  IN_G2_W1};
  const int gB1[3]  = {IN_G0_B1,  IN_G1_B1,  IN_G2_B1};
  const int gW2[3]  = {IN_G0_W2,  IN_G1_W2,  IN_G2_W2};
  const int gB2[3]  = {IN_G0_B2,  IN_G1_B2,  IN_G2_B2};
  const int gEps[3] = {IN_G0_EPS, IN_G1_EPS, IN_G2_EPS};
  for (int L = 0; L < 3; ++L) {
    prep(gW1[L], gB1[L], -1, -1, wtG1[L], bG1[L], 64, 128, 64, 128);
    prep(gW2[L], gB2[L], -1, -1, wtG2[L], bG2[L], 128, 64, 128, 64);
  }
  prep(IN_C0_W, IN_C0_B, IN_C0_GAMMA, IN_C0_BETA, wtC0, bC0, 144, 128, 160, 128);
  prep(IN_C1_W, IN_C1_B, IN_C1_GAMMA, IN_C1_BETA, wtC1, bC1, 128, 64, 128, 64);
  prep(IN_C2_W, IN_C2_B, -1, -1, wtC2, bC2, 64, 2, 64, 16);

  // ---- node encoder ----
  enc_kernel<<<N_NODES / 16, 256, 0, stream>>>(nf, wtEnc, bEnc, x);

  // ---- GIN layers (last layer also writes the f16 shadow of x) ----
  const int nXf = N_NODES * HDIM;
  for (int L = 0; L < 3; ++L) {
    zero_kernel<<<(nXf + 255) / 256, 256, 0, stream>>>(agg, nXf);
    scatter_kernel<<<(N_EDGES * 16) / 256, 256, 0, stream>>>(x, src, dst, agg);
    gin_mlp_kernel<<<N_NODES / 16, 256, 0, stream>>>(
        x, (L == 2) ? x16 : nullptr, agg, (const float*)d_in[gEps[L]],
        wtG1[L], bG1[L], wtG2[L], bG2[L]);
  }

  // ---- edge classifier: 64 edges per block ----
  edge_cls_kernel<<<N_EDGES / 64, 256, 0, stream>>>(
      x16, eattr, src, dst, wtC0, bC0, wtC1, bC1, wtC2, bC2, (float*)d_out);
}
